// TypeLayer_Extend_56349970923981
// MI455X (gfx1250) — compile-verified, matrix operand order checked by reading, not run on hardware
//
#include <hip/hip_runtime.h>

// ---------------------------------------------------------------------------
// MI455X (gfx1250, wave32). All four GEMMs on v_wmma_f32_16x16x32_bf16.
// Operands pre-converted to bf16 once so LDS tiles are staged with
// GLOBAL_LOAD_ASYNC_TO_LDS_B128 (ASYNCcnt) when the toolchain exposes it.
// prob-GEMM fused with candidate mask + inverse-CDF categorical sampling +
// atomic scatter (prob never touches HBM).
// ---------------------------------------------------------------------------

typedef __attribute__((ext_vector_type(16))) __bf16 v16bf;
typedef __attribute__((ext_vector_type(8)))  __bf16 bf16x8;
typedef __attribute__((ext_vector_type(8)))  float  v8f;

constexpr int NB = 8, NN = 256, ND = 256, NO = 256, NR = 512, NE = 8192;

#if defined(__has_builtin)
#  if __has_builtin(__builtin_amdgcn_global_load_async_to_lds_b128) && \
      __has_builtin(__builtin_amdgcn_s_wait_asynccnt)
#    define USE_ASYNC_LDS 1
#  endif
#endif
#ifndef USE_ASYNC_LDS
#  define USE_ASYNC_LDS 0
#endif

#if USE_ASYNC_LDS
// Match the builtin's parameter types exactly (from the compiler diagnostic):
// (int4 addrspace(1)*, int4 addrspace(3)*, imm offset, imm cpol)
typedef int v4i __attribute__((__vector_size__(4 * sizeof(int))));
typedef __attribute__((address_space(1))) v4i* gv4i_p;
typedef __attribute__((address_space(3))) v4i* lv4i_p;
#endif

// 16-byte global -> LDS copy. Async path writes LDS directly (no VGPR data),
// tracked by ASYNCcnt; fallback is a plain vector copy.
__device__ __forceinline__ void cp16(void* lds, const void* glob) {
#if USE_ASYNC_LDS
  __builtin_amdgcn_global_load_async_to_lds_b128(
      (gv4i_p)(uintptr_t)glob,
      (lv4i_p)(uint32_t)(uintptr_t)lds,
      0, 0);
#else
  *(uint4*)lds = *(const uint4*)glob;
#endif
}
__device__ __forceinline__ void cp_wait() {
#if USE_ASYNC_LDS
  __builtin_amdgcn_s_wait_asynccnt(0);
#endif
}

// A/B fragment for 16x16x32 bf16 WMMA: lane holds K = kbase..kbase+7 and
// kbase+16..kbase+23 of its row (kbase = (lane>>4)*8). Two 16B LDS loads.
__device__ __forceinline__ v16bf load_frag(const __bf16* row, int kbase) {
  bf16x8 lo = *(const bf16x8*)(row + kbase);
  bf16x8 hi = *(const bf16x8*)(row + kbase + 16);
  v16bf f;
#pragma unroll
  for (int i = 0; i < 8; ++i) { f[i] = lo[i]; f[8 + i] = hi[i]; }
  return f;
}

__device__ __forceinline__ v8f wmma_bf16(v16bf a, v16bf b, v8f c) {
  return __builtin_amdgcn_wmma_f32_16x16x32_bf16(false, a, false, b,
                                                 (short)0, c, false, false);
}

__device__ __forceinline__ float rng01(unsigned x) {
  x ^= x >> 17; x *= 0xed5ad4bbu;
  x ^= x >> 11; x *= 0xac4c1b51u;
  x ^= x >> 15; x *= 0x31848babu;
  x ^= x >> 14;
  return (float)(x >> 8) * (1.0f / 16777216.0f);   // [0,1)
}

// ---------------------------------------------------------------------------
// bf16 conversion pre-passes (enable byte-exact async LDS staging).
// ---------------------------------------------------------------------------
__global__ __launch_bounds__(256) void k_cvt(const float* __restrict__ s,
                                             __bf16* __restrict__ d, int n) {
  int i = blockIdx.x * blockDim.x + threadIdx.x;
  int stride = gridDim.x * blockDim.x;
  for (; i < n; i += stride) d[i] = (__bf16)s[i];
}

// WkT[o][d] = Wk[d][o]
__global__ __launch_bounds__(256) void k_cvtT(const float* __restrict__ Wk,
                                              __bf16* __restrict__ WkT) {
  int d = blockIdx.x;
  int o = threadIdx.x;
  WkT[(size_t)o * ND + d] = (__bf16)Wk[(size_t)d * NO + o];
}

// ---------------------------------------------------------------------------
// Kernel: G[b] = Xh[b] @ Xh[b]^T.  block = (b, 64-row strip); 8 waves, each
// one 16-row tile x eight 16-col tiles; K staged in 32-wide bf16 LDS chunks.
// ---------------------------------------------------------------------------
__global__ __launch_bounds__(256) void k_gram(const __bf16* __restrict__ Xh,
                                              float* __restrict__ G) {
  __shared__ __align__(16) __bf16 Abuf[64][264];
  __shared__ __align__(16) __bf16 Bbuf[256][40];
  const int t = threadIdx.x;
  const int b = blockIdx.x >> 2;
  const int rb = blockIdx.x & 3;
  const int lane = t & 31, wave = t >> 5;
  const int rt = wave & 3, ch = wave >> 2;
  const int kb_lane = (lane >> 4) * 8;
  const int arow = rt * 16 + (lane & 15);

  // A strip: 64 rows x 256 K = 32 x 16B chunks per row
  for (int idx = t; idx < 64 * 32; idx += 256) {
    int r = idx >> 5, c = idx & 31;
    cp16(&Abuf[r][c * 8], Xh + ((size_t)(b * NN + rb * 64 + r)) * ND + c * 8);
  }

  v8f acc[8];
#pragma unroll
  for (int j = 0; j < 8; ++j)
#pragma unroll
    for (int v = 0; v < 8; ++v) acc[j][v] = 0.f;

  for (int kb = 0; kb < ND / 32; ++kb) {
    __syncthreads();
    for (int idx = t; idx < 256 * 4; idx += 256) {       // 256 rows x 64B
      int m = idx >> 2, c = idx & 3;
      cp16(&Bbuf[m][c * 8], Xh + ((size_t)(b * NN + m)) * ND + kb * 32 + c * 8);
    }
    cp_wait();
    __syncthreads();
    v16bf a = load_frag(&Abuf[arow][kb * 32], kb_lane);
    v16bf bb[8];
#pragma unroll
    for (int j = 0; j < 8; ++j)
      bb[j] = load_frag(&Bbuf[(ch * 8 + j) * 16 + (lane & 15)][0], kb_lane);
#pragma unroll
    for (int j = 0; j < 8; ++j) acc[j] = wmma_bf16(a, bb[j], acc[j]);
  }

#pragma unroll
  for (int j = 0; j < 8; ++j) {
    int col = (ch * 8 + j) * 16 + (lane & 15);
#pragma unroll
    for (int v = 0; v < 8; ++v) {
      int row = rb * 64 + rt * 16 + v + ((lane >> 4) << 3);
      G[((size_t)(b * NN + row)) * NN + col] = acc[j][v];
    }
  }
}

// ---------------------------------------------------------------------------
// sim row softmax (logit max is exactly 0 at the diagonal) + inclusive CDF.
// sim emitted directly in bf16 (GEMM A-operand of the fused edge kernel).
// ---------------------------------------------------------------------------
__global__ __launch_bounds__(256) void k_sim(const float* __restrict__ G,
                                             __bf16* __restrict__ simh,
                                             float* __restrict__ cdf) {
  __shared__ float sc[256];
  const int t = threadIdx.x;
  const int b = blockIdx.x >> 8;
  const int n = blockIdx.x & 255;
  const float* Gr = G + ((size_t)(b * NN + n)) * NN;
  float g  = Gr[t];
  float dn = Gr[n];
  float dm = G[((size_t)(b * NN + t)) * NN + t];
  float d2 = fmaxf(dn + dm - 2.f * g, 0.f);
  float e  = __expf(-sqrtf(d2));
  sc[t] = e;
  __syncthreads();
  for (int off = 1; off < 256; off <<= 1) {        // Hillis-Steele scan
    float v = (t >= off) ? sc[t - off] : 0.f;
    __syncthreads();
    sc[t] += v;
    __syncthreads();
  }
  float inv = 1.f / sc[255];
  simh[((size_t)(b * NN + n)) * NN + t] = (__bf16)(e * inv);
  cdf [((size_t)(b * NN + n)) * NN + t] = sc[t] * inv;
}

__global__ void k_zero(float* __restrict__ p, int n) {
  int i = blockIdx.x * blockDim.x + threadIdx.x;
  int stride = gridDim.x * blockDim.x;
  for (; i < n; i += stride) p[i] = 0.f;
}

__global__ __launch_bounds__(256) void k_base(const int* __restrict__ heads,
                                              const int* __restrict__ rels,
                                              const int* __restrict__ tails,
                                              float* __restrict__ counts) {
  int i = blockIdx.x * 256 + threadIdx.x;
  if (i >= NB * NE) return;
  int b = i / NE;
  int h = heads[i], r = rels[i], tl = tails[i];
  atomicAdd(&counts[((size_t)(b * NN + h)) * NR + r], 1.f);
  atomicAdd(&counts[((size_t)(b * NN + tl)) * NR + r], 1.f);
}

// ---------------------------------------------------------------------------
// Fused: prob = sim @ score^T (WMMA) -> candidate mask -> inverse-CDF draw
// -> atomic scatter into counts.
// ---------------------------------------------------------------------------
__global__ __launch_bounds__(256) void k_edge(const __bf16* __restrict__ simh,
                                              const __bf16* __restrict__ Sh,
                                              const float* __restrict__ cdf,
                                              const int* __restrict__ edge_mat,
                                              const int* __restrict__ epsp,
                                              float* __restrict__ counts) {
  __shared__ __align__(16) __bf16 Abuf[64][264];
  __shared__ __align__(16) __bf16 Bbuf[256][40];
  const int t = threadIdx.x;
  const int b = blockIdx.x >> 2;
  const int rb = blockIdx.x & 3;
  const int lane = t & 31, wave = t >> 5;
  const int rt = wave & 3, ch = wave >> 2;
  const int kb_lane = (lane >> 4) * 8;
  const int arow = rt * 16 + (lane & 15);

  for (int idx = t; idx < 64 * 32; idx += 256) {
    int r = idx >> 5, c = idx & 31;
    cp16(&Abuf[r][c * 8], simh + ((size_t)(b * NN + rb * 64 + r)) * NN + c * 8);
  }

  v8f acc[8];
#pragma unroll
  for (int j = 0; j < 8; ++j)
#pragma unroll
    for (int v = 0; v < 8; ++v) acc[j][v] = 0.f;

  for (int kb = 0; kb < NN / 32; ++kb) {
    __syncthreads();
    for (int idx = t; idx < 256 * 4; idx += 256) {
      int m = idx >> 2, c = idx & 3;
      cp16(&Bbuf[m][c * 8], Sh + ((size_t)(b * NN + m)) * NN + kb * 32 + c * 8);
    }
    cp_wait();
    __syncthreads();
    v16bf a = load_frag(&Abuf[arow][kb * 32], kb_lane);
    v16bf bb[8];
#pragma unroll
    for (int j = 0; j < 8; ++j)
      bb[j] = load_frag(&Bbuf[(ch * 8 + j) * 16 + (lane & 15)][0], kb_lane);
#pragma unroll
    for (int j = 0; j < 8; ++j) acc[j] = wmma_bf16(a, bb[j], acc[j]);
  }

  const float epsf = (float)epsp[0];
#pragma unroll
  for (int j = 0; j < 8; ++j) {
    const int m = (ch * 8 + j) * 16 + (lane & 15);
#pragma unroll
    for (int v = 0; v < 8; ++v) {
      const int n = rb * 64 + rt * 16 + v + ((lane >> 4) << 3);
      float p = acc[j][v];
      int e0 = edge_mat[((size_t)(b * NN + n)) * NN + m];
      if (p > epsf && e0 < 1) {
        float u = rng01(((unsigned)((b * NN + n) * NN + m)) * 0x9E3779B9u +
                        0x85EBCA6Bu);
        const float* crow = cdf + ((size_t)(b * NN + n)) * NN;
        __builtin_prefetch(crow, 0, 0);             // global_prefetch_b8
        int s = 0;                                  // inverse-CDF draw
#pragma unroll
        for (int st = 128; st > 0; st >>= 1)
          if (crow[s + st - 1] <= u) s += st;
        s = s < 255 ? s : 255;
        int nr = edge_mat[((size_t)(b * NN + s)) * NN + m];
        if (nr != 0) {
          atomicAdd(&counts[((size_t)(b * NN + n)) * NR + nr], 1.f);
          atomicAdd(&counts[((size_t)(b * NN + m)) * NR + nr], 1.f);
        }
      }
    }
  }
}

// ---------------------------------------------------------------------------
// WrT[o][r] = (rel @ Wk + bk)[r][o], emitted in bf16 for k_out's B operand.
// ---------------------------------------------------------------------------
__global__ __launch_bounds__(256) void k_wrt(const __bf16* __restrict__ relh,
                                             const __bf16* __restrict__ WkT,
                                             const float* __restrict__ bk,
                                             __bf16* __restrict__ WrTh) {
  __shared__ __align__(16) __bf16 Abuf[64][264];
  __shared__ __align__(16) __bf16 Bbuf[256][40];
  const int t = threadIdx.x;
  const int rblk = blockIdx.x;                     // NR/64 = 8
  const int lane = t & 31, wave = t >> 5;
  const int rt = wave & 3, ch = wave >> 2;
  const int kb_lane = (lane >> 4) * 8;
  const int arow = rt * 16 + (lane & 15);

  for (int idx = t; idx < 64 * 32; idx += 256) {
    int r = idx >> 5, c = idx & 31;
    cp16(&Abuf[r][c * 8], relh + ((size_t)(rblk * 64 + r)) * ND + c * 8);
  }

  v8f acc[8];
#pragma unroll
  for (int j = 0; j < 8; ++j)
#pragma unroll
    for (int v = 0; v < 8; ++v) acc[j][v] = 0.f;

  for (int kb = 0; kb < ND / 32; ++kb) {
    __syncthreads();
    for (int idx = t; idx < 256 * 4; idx += 256) {
      int o = idx >> 2, c = idx & 3;
      cp16(&Bbuf[o][c * 8], WkT + (size_t)o * ND + kb * 32 + c * 8);
    }
    cp_wait();
    __syncthreads();
    v16bf a = load_frag(&Abuf[arow][kb * 32], kb_lane);
    v16bf bb[8];
#pragma unroll
    for (int j = 0; j < 8; ++j)
      bb[j] = load_frag(&Bbuf[(ch * 8 + j) * 16 + (lane & 15)][0], kb_lane);
#pragma unroll
    for (int j = 0; j < 8; ++j) acc[j] = wmma_bf16(a, bb[j], acc[j]);
  }

#pragma unroll
  for (int j = 0; j < 8; ++j) {
    int o = (ch * 8 + j) * 16 + (lane & 15);
#pragma unroll
    for (int v = 0; v < 8; ++v) {
      int r = rblk * 64 + rt * 16 + v + ((lane >> 4) << 3);
      WrTh[(size_t)o * NR + r] = (__bf16)(acc[j][v] + bk[o]);
    }
  }
}

// ---------------------------------------------------------------------------
// out = relu(counts @ Wr): [B*N,512] x [512,256]. counts staged with
// convert (f32 atomically accumulated), Wr^T staged async (already bf16).
// ---------------------------------------------------------------------------
__global__ __launch_bounds__(256) void k_out(const float* __restrict__ counts,
                                             const __bf16* __restrict__ WrTh,
                                             float* __restrict__ out) {
  __shared__ __align__(16) __bf16 Ak[64][40];
  __shared__ __align__(16) __bf16 Bb[256][40];
  const int t = threadIdx.x;
  const int rowblk = blockIdx.x;                   // (NB*NN)/64 = 32
  const int lane = t & 31, wave = t >> 5;
  const int rt = wave & 3, ch = wave >> 2;
  const int kb_lane = (lane >> 4) * 8;
  const int arow = rt * 16 + (lane & 15);

  v8f acc[8];
#pragma unroll
  for (int j = 0; j < 8; ++j)
#pragma unroll
    for (int v = 0; v < 8; ++v) acc[j][v] = 0.f;

  for (int kb = 0; kb < NR / 32; ++kb) {
    __syncthreads();
    for (int idx = t; idx < 64 * 32; idx += 256) {
      int r = idx >> 5, kk = idx & 31;
      Ak[r][kk] = (__bf16)counts[((size_t)(rowblk * 64 + r)) * NR + kb * 32 + kk];
    }
    for (int idx = t; idx < 256 * 4; idx += 256) {
      int o = idx >> 2, c = idx & 3;
      cp16(&Bb[o][c * 8], WrTh + (size_t)o * NR + kb * 32 + c * 8);
    }
    cp_wait();
    __syncthreads();
    v16bf a = load_frag(&Ak[arow][0], kb_lane);
    v16bf bb[8];
#pragma unroll
    for (int j = 0; j < 8; ++j)
      bb[j] = load_frag(&Bb[(ch * 8 + j) * 16 + (lane & 15)][0], kb_lane);
#pragma unroll
    for (int j = 0; j < 8; ++j) acc[j] = wmma_bf16(a, bb[j], acc[j]);
  }

#pragma unroll
  for (int j = 0; j < 8; ++j) {
    int o = (ch * 8 + j) * 16 + (lane & 15);
#pragma unroll
    for (int v = 0; v < 8; ++v) {
      int row = rowblk * 64 + rt * 16 + v + ((lane >> 4) << 3);
      out[(size_t)row * NO + o] = fmaxf(acc[j][v], 0.f);
    }
  }
}

// ---------------------------------------------------------------------------
extern "C" void kernel_launch(void* const* d_in, const int* in_sizes, int n_in,
                              void* d_out, int out_size, void* d_ws,
                              size_t ws_size, hipStream_t stream) {
  (void)in_sizes; (void)n_in; (void)out_size; (void)ws_size;
  const float* X     = (const float*)d_in[0];   // [B,N,D]
  const float* score = (const float*)d_in[1];   // [B,N,N]
  const float* rel   = (const float*)d_in[2];   // [R,D]
  const float* Wk    = (const float*)d_in[3];   // [D,O]
  const float* bk    = (const float*)d_in[4];   // [O]
  const int* heads   = (const int*)d_in[6];
  const int* rels    = (const int*)d_in[7];
  const int* tails   = (const int*)d_in[8];
  const int* edge    = (const int*)d_in[9];
  const int* epsp    = (const int*)d_in[10];

  char* ws = (char*)d_ws;
  float*  G    = (float*)(ws);                                  // 2 MB
  float*  cdf  = (float*)(ws + (2u << 20));                     // 2 MB
  float*  cnt  = (float*)(ws + (4u << 20));                     // 4 MB
  __bf16* Xh   = (__bf16*)(ws + (8u << 20));                    // 1 MB
  __bf16* Sh   = (__bf16*)(ws + (9u << 20));                    // 1 MB
  __bf16* simh = (__bf16*)(ws + (10u << 20));                   // 1 MB
  __bf16* relh = (__bf16*)(ws + (11u << 20));                   // 256 KB
  __bf16* WkT  = (__bf16*)(ws + (11u << 20) + (256u << 10));    // 128 KB
  __bf16* WrTh = (__bf16*)(ws + (11u << 20) + (384u << 10));    // 256 KB
  float*  out  = (float*)d_out;

  k_cvt <<<512, 256, 0, stream>>>(X, Xh, NB * NN * ND);
  k_cvt <<<512, 256, 0, stream>>>(score, Sh, NB * NN * NN);
  k_cvt <<<256, 256, 0, stream>>>(rel, relh, NR * ND);
  k_cvtT<<<ND,  256, 0, stream>>>(Wk, WkT);
  k_gram<<<NB * (NN / 64), 256, 0, stream>>>(Xh, G);
  k_sim <<<NB * NN,        256, 0, stream>>>(G, simh, cdf);
  k_zero<<<1024,           256, 0, stream>>>(cnt, NB * NN * NR);
  k_base<<<(NB * NE) / 256,256, 0, stream>>>(heads, rels, tails, cnt);
  k_edge<<<NB * (NN / 64), 256, 0, stream>>>(simh, Sh, cdf, edge, epsp, cnt);
  k_wrt <<<NR / 64,        256, 0, stream>>>(relh, WkT, bk, WrTh);
  k_out <<<(NB * NN) / 64, 256, 0, stream>>>(cnt, WrTh, out);
}
